// Mymodel_51161650430330
// MI455X (gfx1250) — matrix operand order repaired and checked by reading, not compiled, not run on hardware
//
#include <hip/hip_runtime.h>
#include <hip/hip_bf16.h>

// ---------------- problem constants ----------------
#define S_LEN   2048
#define INDIM   64
#define OUTDIM  8
#define NU      1024
#define NL      4
#define TSYN    64
#define KCAT    (2*NU)          // packed K: [input(1024 padded) | hidden(1024)]
#define ROWS    (4*NU)          // 4 gates x 1024 units
#define PARTS   16              // workgroups per layer
#define UPW     (NU/PARTS)      // 64 units per workgroup
#define TPB     512             // 16 waves (wave32): one 16-row tile per wave
#define OFFSET_C (-6.0f)

// ---------------- workspace layout (bytes) ----------------
#define WCAT_B  ((size_t)NL*ROWS*KCAT*2)     // 64 MB bf16 packed weights
#define HBUF_B  ((size_t)NL*S_LEN*NU*2)      // 16 MB bf16 hidden states
#define FLAG_B  ((size_t)NL*S_LEN*4)         // 32 KB sync flags
#define ACT_B   ((size_t)S_LEN*OUTDIM*4)     // softplus activations
#define DEV_B   ((size_t)S_LEN*4*4)          // conv output

typedef __attribute__((ext_vector_type(16))) __bf16        v16bf;
typedef __attribute__((ext_vector_type(8)))  float         v8f;
typedef __attribute__((ext_vector_type(4)))  unsigned int  u32x4;

union ABPack { u32x4 q[2]; v16bf v; };

__device__ __forceinline__ unsigned short f2bf(float f) {
  union { float f; unsigned int u; } v; v.f = f;
  unsigned int u = v.u;
  unsigned int r = (u + 0x7FFFu + ((u >> 16) & 1u)) >> 16;  // RNE
  return (unsigned short)r;
}
__device__ __forceinline__ float bf2f(unsigned short h) {
  union { float f; unsigned int u; } v; v.u = ((unsigned int)h) << 16;
  return v.f;
}
__device__ __forceinline__ float sigm(float x) { return 1.0f / (1.0f + expf(-x)); }

// A 16x32 bf16 per ISA layout: lane half hi -> K+8; vector elems 8..15 -> K+16
__device__ __forceinline__ void loadA(ABPack& a, const unsigned short* wrow,
                                      int kc, int hi) {
  a.q[0] = *(const u32x4*)(wrow + kc * 32 + hi * 8);
  a.q[1] = *(const u32x4*)(wrow + kc * 32 + 16 + hi * 8);
}
// B 32x16 bf16: broadcast GEMV vector into all 16 columns (uniform LDS read)
__device__ __forceinline__ void loadB(ABPack& b, const unsigned short* vec,
                                      int kc, int hi) {
  const unsigned short* vp = vec + kc * 32 + hi * 8;
  b.q[0] = *(const u32x4*)(vp);
  b.q[1] = *(const u32x4*)(vp + 16);
}

#define WMMA_BF16(acc, a, b) \
  __builtin_amdgcn_wmma_f32_16x16x32_bf16(false, (a).v, false, (b).v, \
                                          (short)0, (acc), false, false)

// No-copy 4-slot software pipeline, unroll x4 (segment length % 4 == 0).
// Each WMMA consumes slot i, then slot i is reloaded in place for kc+4+i:
// no rotation moves, load-to-use distance = 3 WMMAs, ~6 load-pairs in flight.
// Slot preloads may run <=448B past the segment/row end; that remains inside
// the contiguous workspace and the values are never consumed.
__device__ __forceinline__ void gemv_seg4(const unsigned short* __restrict__ wrow,
                                          const unsigned short* __restrict__ vec,
                                          int lo, int hi_kc, int hi, v8f& acc) {
  ABPack a0, b0, a1, b1, a2, b2, a3, b3;
  loadA(a0, wrow, lo + 0, hi); loadB(b0, vec, lo + 0, hi);
  loadA(a1, wrow, lo + 1, hi); loadB(b1, vec, lo + 1, hi);
  loadA(a2, wrow, lo + 2, hi); loadB(b2, vec, lo + 2, hi);
  loadA(a3, wrow, lo + 3, hi); loadB(b3, vec, lo + 3, hi);
  for (int kc = lo; kc < hi_kc; kc += 4) {
    __builtin_prefetch((const void*)(wrow + (kc + 12) * 32 + hi * 8), 0, 3);
    acc = WMMA_BF16(acc, a0, b0);
    loadA(a0, wrow, kc + 4, hi); loadB(b0, vec, kc + 4, hi);
    acc = WMMA_BF16(acc, a1, b1);
    loadA(a1, wrow, kc + 5, hi); loadB(b1, vec, kc + 5, hi);
    acc = WMMA_BF16(acc, a2, b2);
    loadA(a2, wrow, kc + 6, hi); loadB(b2, vec, kc + 6, hi);
    acc = WMMA_BF16(acc, a3, b3);
    loadA(a3, wrow, kc + 7, hi); loadB(b3, vec, kc + 7, hi);
  }
}

// Layer-0 input half: exactly 2 K-chunks (x is 64-wide) -> trivial unrolled.
__device__ __forceinline__ void gemv_2chunk(const unsigned short* __restrict__ wrow,
                                            const unsigned short* __restrict__ vec,
                                            int hi, v8f& acc) {
  ABPack a0, b0, a1, b1;
  loadA(a0, wrow, 0, hi); loadB(b0, vec, 0, hi);
  loadA(a1, wrow, 1, hi); loadB(b1, vec, 1, hi);
  acc = WMMA_BF16(acc, a0, b0);
  acc = WMMA_BF16(acc, a1, b1);
}

// ---------------- flag reset (graph replay must be deterministic) ----------
__global__ void k_zero_flags(int* flags) {
  int i = blockIdx.x * blockDim.x + threadIdx.x;
  if (i < NL * S_LEN) flags[i] = 0;
}

// ---------------- pack f32 weights -> bf16 W_cat[l][4096][2048] ------------
__global__ void k_pack_weights(const float* __restrict__ Wih0,
                               const float* __restrict__ Wih,
                               const float* __restrict__ Whh,
                               unsigned short* __restrict__ wcat) {
  size_t idx = (size_t)blockIdx.x * blockDim.x + threadIdx.x;  // NL*ROWS*KCAT = 2^25
  int l   = (int)(idx >> 23);                 // ROWS*KCAT = 2^23
  int rem = (int)(idx & ((1u << 23) - 1));
  int row = rem >> 11;                        // KCAT = 2^11
  int col = rem & (KCAT - 1);
  float v;
  if (col < NU) {                             // input-weight half
    if (l == 0) v = (col < INDIM) ? Wih0[row * INDIM + col] : 0.0f;
    else        v = Wih[(size_t)((l - 1) * ROWS + row) * NU + col];
  } else {                                    // hidden-weight half
    v = Whh[(size_t)(l * ROWS + row) * NU + (col - NU)];
  }
  wcat[idx] = f2bf(v);
}

// ---------------- persistent pipelined LSTM (64 WGs = 4 layers x 16 parts) -
__global__ void __launch_bounds__(TPB, 1)
k_lstm(const float* __restrict__ x,
       const float* __restrict__ bias,
       const unsigned short* __restrict__ wcat,
       unsigned short* __restrict__ hbuf,
       int* __restrict__ flags) {
  __shared__ __attribute__((aligned(16))) unsigned short vec[KCAT]; // bf16 [inp|h]
  __shared__ float zbuf[4 * UPW];   // i,f,g,o pre-activations for our 64 units
  __shared__ float cbuf[UPW];       // cell state slice (f32, persistent)

  const int layer = blockIdx.x >> 4;
  const int part  = blockIdx.x & 15;
  const int tid   = threadIdx.x;
  const int lane  = tid & 31;
  const int wave  = tid >> 5;       // 0..15: one 16-row tile per wave
  const int m     = lane & 15;      // output row within 16-row tile
  const int hi    = lane >> 4;      // half-wave selector (ISA K split)

  // tile -> (gate, subtile) mapping; weight row for this lane
  const int gate = wave >> 2;
  const int sub  = wave & 3;
  const int row  = gate * NU + part * UPW + sub * 16 + m;
  const unsigned short* wrow = wcat + ((size_t)(layer * ROWS + row)) * KCAT;

  if (tid < UPW) cbuf[tid] = 0.0f;
  __syncthreads();

  for (int t = 0; t < S_LEN; ++t) {
    // ---- wait for producers: layer-1 @ t, and full self-h @ t-1 ----
    if (tid == 0) {
      if (layer > 0) {
        const int* f = &flags[(layer - 1) * S_LEN + t];
        while (__hip_atomic_load(f, __ATOMIC_ACQUIRE, __HIP_MEMORY_SCOPE_AGENT) < PARTS)
          __builtin_amdgcn_s_sleep(1);
      }
      if (t > 0) {
        const int* f = &flags[layer * S_LEN + (t - 1)];
        while (__hip_atomic_load(f, __ATOMIC_ACQUIRE, __HIP_MEMORY_SCOPE_AGENT) < PARTS)
          __builtin_amdgcn_s_sleep(1);
      }
    }
    __syncthreads();

    // ---- stage combined input vector into LDS (bf16) ----
    for (int i = tid; i < KCAT; i += TPB) {
      unsigned short hv;
      if (i < NU) {
        if (layer == 0) hv = (i < INDIM) ? f2bf(x[t * INDIM + i]) : (unsigned short)0;
        else            hv = hbuf[((size_t)((layer - 1) * S_LEN + t)) * NU + i];
      } else {
        hv = (t == 0) ? (unsigned short)0
                      : hbuf[((size_t)(layer * S_LEN + (t - 1))) * NU + (i - NU)];
      }
      vec[i] = hv;
    }
    __syncthreads();

    // ---- GEMV via WMMA over two uniform K segments ----
    v8f acc = {0.f, 0.f, 0.f, 0.f, 0.f, 0.f, 0.f, 0.f};
    if (layer == 0) gemv_2chunk(wrow, vec, hi, acc);          // x half (2 chunks)
    else            gemv_seg4(wrow, vec, 0, NU / 32, hi, acc);// input-weight half
    gemv_seg4(wrow, vec, NU / 32, KCAT / 32, hi, acc);        // hidden-weight half

    // D col 0 == GEMV result; lanes 0 (M=rr) and 16 (M=8+rr) hold it
    if (m == 0) {
      const int ub = sub * 16 + hi * 8;
      #pragma unroll
      for (int rr = 0; rr < 8; ++rr) zbuf[gate * UPW + ub + rr] = acc[rr];
    }
    __syncthreads();

    // ---- LSTM gates (f32), update c, emit h (bf16) ----
    if (tid < UPW) {
      const int u  = tid;
      const int bb = layer * ROWS + part * UPW + u;
      float zi = zbuf[0 * UPW + u] + bias[bb + 0 * NU];
      float zf = zbuf[1 * UPW + u] + bias[bb + 1 * NU];
      float zg = zbuf[2 * UPW + u] + bias[bb + 2 * NU];
      float zo = zbuf[3 * UPW + u] + bias[bb + 3 * NU];
      float c  = sigm(zf) * cbuf[u] + sigm(zi) * tanhf(zg);
      float h  = sigm(zo) * tanhf(c);
      cbuf[u] = c;
      hbuf[((size_t)(layer * S_LEN + t)) * NU + part * UPW + u] = f2bf(h);
    }
    __threadfence();
    __syncthreads();
    if (tid == 0)
      __hip_atomic_fetch_add(&flags[layer * S_LEN + t], 1,
                             __ATOMIC_RELEASE, __HIP_MEMORY_SCOPE_AGENT);
  }
}

// ---------------- FC + softplus head ----------------
__global__ void k_fc(const unsigned short* __restrict__ hbuf,
                     const float* __restrict__ Wfc,
                     const float* __restrict__ bfc,
                     float* __restrict__ act) {
  int t = blockIdx.x * blockDim.x + threadIdx.x;
  if (t >= S_LEN) return;
  const unsigned short* hrow = hbuf + ((size_t)((NL - 1) * S_LEN + t)) * NU;
  float y[OUTDIM];
  #pragma unroll
  for (int o = 0; o < OUTDIM; ++o) y[o] = 0.f;
  for (int k = 0; k < NU; ++k) {
    float hv = bf2f(hrow[k]);
    #pragma unroll
    for (int o = 0; o < OUTDIM; ++o) y[o] += hv * Wfc[o * NU + k];
  }
  #pragma unroll
  for (int o = 0; o < OUTDIM; ++o) {
    float v  = y[o] + bfc[o] + OFFSET_C;
    float sp = v > 0.f ? v + log1pf(expf(-v)) : log1pf(expf(v));  // softplus
    act[t * OUTDIM + o] = sp;
  }
}

// ---------------- causal conv: dev[t,j] = sum_d act[t-d] @ syn[:,d,j] -------
__global__ void k_conv(const float* __restrict__ act,
                       const float* __restrict__ syn,   // [OUT][TSYN][4]
                       float* __restrict__ dev) {
  int idx = blockIdx.x * blockDim.x + threadIdx.x;   // S_LEN*4
  if (idx >= S_LEN * 4) return;
  int t = idx >> 2, j = idx & 3;
  int dmax = t < (TSYN - 1) ? t : (TSYN - 1);
  float s = 0.f;
  for (int d = 0; d <= dmax; ++d) {
    const float* a = act + (size_t)(t - d) * OUTDIM;
    #pragma unroll
    for (int o = 0; o < OUTDIM; ++o) s += a[o] * syn[(o * TSYN + d) * 4 + j];
  }
  dev[idx] = s;
}

// ---------------- cumulative position ----------------
__global__ void k_cumsum(const float* __restrict__ x,
                         const float* __restrict__ dev,
                         float* __restrict__ out) {
  int j = threadIdx.x;
  if (j < 2) {
    float p0 = x[j];        // x[0,0,:2]
    float run = 0.f;
    for (int t = 0; t < S_LEN; ++t) {
      run += dev[t * 4 + j] - dev[t * 4 + 2 + j];
      out[t * 2 + j] = p0 + run;
    }
  }
}

// ---------------- host entry ----------------
extern "C" void kernel_launch(void* const* d_in, const int* in_sizes, int n_in,
                              void* d_out, int out_size, void* d_ws, size_t ws_size,
                              hipStream_t stream) {
  (void)in_sizes; (void)n_in; (void)out_size; (void)ws_size;
  const float* x    = (const float*)d_in[0];
  const float* Wih0 = (const float*)d_in[1];
  const float* Wih  = (const float*)d_in[2];
  const float* Whh  = (const float*)d_in[3];
  const float* b    = (const float*)d_in[4];
  const float* Wfc  = (const float*)d_in[5];
  const float* bfc  = (const float*)d_in[6];
  const float* syn  = (const float*)d_in[7];

  char* ws = (char*)d_ws;
  unsigned short* wcat  = (unsigned short*)(ws);
  unsigned short* hbuf  = (unsigned short*)(ws + WCAT_B);
  int*            flags = (int*)(ws + WCAT_B + HBUF_B);
  float*          act   = (float*)(ws + WCAT_B + HBUF_B + FLAG_B);
  float*          dev   = (float*)(ws + WCAT_B + HBUF_B + FLAG_B + ACT_B);
  float*          out   = (float*)d_out;

  k_zero_flags  <<<(NL * S_LEN + 255) / 256, 256, 0, stream>>>(flags);
  k_pack_weights<<<(unsigned)((size_t)NL * ROWS * KCAT / 256), 256, 0, stream>>>(Wih0, Wih, Whh, wcat);
  k_lstm        <<<NL * PARTS, TPB, 0, stream>>>(x, b, wcat, hbuf, flags);
  k_fc          <<<S_LEN / 64, 64, 0, stream>>>(hbuf, Wfc, bfc, act);
  k_conv        <<<(S_LEN * 4) / 256, 256, 0, stream>>>(act, syn, dev);
  k_cumsum      <<<1, 32, 0, stream>>>(x, dev, out);
}
